// Model_4_51659866636675
// MI455X (gfx1250) — matrix operand order repaired and checked by reading, not compiled, not run on hardware
//
#include <hip/hip_runtime.h>
#include <hip/hip_bf16.h>

typedef __attribute__((ext_vector_type(16))) _Float16 v16h;
typedef __attribute__((ext_vector_type(8)))  float    v8f;

#define N_NODES  50000
#define N_EDGES  800000
#define N_GRAPHS 256
#define HID      64
#define SEQ_L    200

static __device__ __forceinline__ v8f wmma_f16(v16h a, v16h b, v8f c) {
  // D = A(16x32 f16) * B(32x16 f16) + C(16x16 f32)
  return __builtin_amdgcn_wmma_f32_16x16x32_f16(false, a, false, b, (short)0, c,
                                                false, false);
}
static __device__ __forceinline__ v8f vzero8() {
  v8f z = {0.f, 0.f, 0.f, 0.f, 0.f, 0.f, 0.f, 0.f};
  return z;
}

// ---------------------------------------------------------------- degree / counts
__global__ void deg_kernel(const int* __restrict__ dst, float* __restrict__ deg, int nE) {
  int e = blockIdx.x * blockDim.x + threadIdx.x;
  if (e < nE) atomicAdd(&deg[dst[e]], 1.0f);
}
__global__ void cnt_kernel(const int* __restrict__ batch, float* __restrict__ cnt, int n) {
  int i = blockIdx.x * blockDim.x + threadIdx.x;
  if (i < n) atomicAdd(&cnt[batch[i]], 1.0f);
}

// ---------------------------------------------------------------- edge scatter (sum x[src] into agg[dst])
__global__ void scatter_kernel(const float* __restrict__ x, const int* __restrict__ src,
                               const int* __restrict__ dst, float* __restrict__ agg, int nE) {
  int e = blockIdx.x * blockDim.x + threadIdx.x;
  if (e >= nE) return;
  const float4* xs = (const float4*)(x + (size_t)src[e] * 64);
  __builtin_prefetch(xs, 0, 3);  // global_prefetch_b8: overlap gather with dst index load
  float* ad = agg + (size_t)dst[e] * 64;
#pragma unroll
  for (int c = 0; c < 16; ++c) {
    float4 v = xs[c];
    atomicAdd(ad + 4 * c + 0, v.x);
    atomicAdd(ad + 4 * c + 1, v.y);
    atomicAdd(ad + 4 * c + 2, v.z);
    atomicAdd(ad + 4 * c + 3, v.w);
  }
}

// ---------------------------------------------------------------- GraphConv node update (WMMA)
// out = relu?( (agg/deg) @ Wrel^T + brel + x @ Wroot^T )
// weights staged per-block to LDS as f16 (8 waves share); one wave per 16-node tile,
// A-fragments loaded once and reused across 4 col tiles (16 WMMA/wave).
__global__ void gconv_kernel(const float* __restrict__ xin, const float* __restrict__ agg,
                             const float* __restrict__ deg,
                             const float* __restrict__ Wrel, const float* __restrict__ brel,
                             const float* __restrict__ Wroot, float* __restrict__ out,
                             int do_relu) {
  __shared__ _Float16 WrS[64 * 64];  // 8 KB
  __shared__ _Float16 WtS[64 * 64];  // 8 KB
  __shared__ float brS[64];

  // cooperative staging (all threads, before any wave exits)
  for (int i = threadIdx.x; i < 64 * 64; i += blockDim.x) {
    WrS[i] = (_Float16)Wrel[i];
    WtS[i] = (_Float16)Wroot[i];
  }
  if (threadIdx.x < 64) brS[threadIdx.x] = brel[threadIdx.x];
  __syncthreads();

  int wave = (blockIdx.x * blockDim.x + threadIdx.x) >> 5;
  int lane = threadIdx.x & 31;
  const int total = N_NODES / 16;  // 3125 node tiles
  if (wave >= total) return;       // wave-uniform exit: EXEC all-ones at every WMMA
  int row0 = wave * 16;
  int m = lane & 15, half = lane >> 4;
  int grow = row0 + m;
  float dinv = __builtin_amdgcn_rcpf(fmaxf(deg[grow], 1.0f));

  // A fragments (agg/deg and x), loaded once per wave
  v16h aR[2], aT[2];
#pragma unroll
  for (int kc = 0; kc < 2; ++kc) {
#pragma unroll
    for (int i = 0; i < 16; ++i) {
      int k = kc * 32 + half * 16 + i;
      aR[kc][i] = (_Float16)(agg[(size_t)grow * 64 + k] * dinv);
      aT[kc][i] = (_Float16)(xin[(size_t)grow * 64 + k]);
    }
  }

#pragma unroll
  for (int ct = 0; ct < 4; ++ct) {
    int n = ct * 16 + (lane & 15);   // output column / weight row
    v8f accR = vzero8(), accT = vzero8();
#pragma unroll
    for (int kc = 0; kc < 2; ++kc) {
      int kb = kc * 32 + half * 16;
      v16h bR, bT;
#pragma unroll
      for (int i = 0; i < 16; ++i) {
        bR[i] = WrS[n * 64 + kb + i];
        bT[i] = WtS[n * 64 + kb + i];
      }
      accR = wmma_f16(aR[kc], bR, accR);
      accT = wmma_f16(aT[kc], bT, accT);
    }
    float bias = brS[n];
#pragma unroll
    for (int r = 0; r < 8; ++r) {
      float v = accR[r] + accT[r] + bias;
      if (do_relu) v = fmaxf(v, 0.0f);
      out[(size_t)(row0 + r + 8 * half) * 64 + n] = v;
    }
  }
}

// ---------------------------------------------------------------- global mean pool
__global__ void pool_scatter_kernel(const float* __restrict__ h, const int* __restrict__ batch,
                                    float* __restrict__ pooled, int n) {
  int i = blockIdx.x * blockDim.x + threadIdx.x;
  if (i >= n) return;
  const float* hr = h + (size_t)i * 64;
  float* pr = pooled + (size_t)batch[i] * 64;
#pragma unroll 4
  for (int c = 0; c < 64; ++c) atomicAdd(pr + c, hr[c]);
}
__global__ void pool_div_kernel(float* pooled, const float* __restrict__ cnt) {
  int i = blockIdx.x * blockDim.x + threadIdx.x;  // 256*64 threads
  pooled[i] /= fmaxf(cnt[i >> 6], 1.0f);
}

// ---------------------------------------------------------------- persistent GRU (1 block, 32 waves)
// h in LDS as f16 (WMMA A-frags) + f32 master in registers; W_ih/W_hh staged ONCE to LDS as f16.
// wave w: graph-block gb=w>>1 (16 graphs), column groups jb = {2*(w&1), 2*(w&1)+1}.
__global__ void __launch_bounds__(1024, 1)
gru_kernel(const int* __restrict__ seq, const int* __restrict__ lengths,
           const float* __restrict__ emb,
           const float* __restrict__ Wih, const float* __restrict__ Whh,
           const float* __restrict__ bih, const float* __restrict__ bhh,
           float* __restrict__ hT) {
  __shared__ _Float16 WihS[192 * 64];      // 24 KB
  __shared__ _Float16 WhhS[192 * 64];      // 24 KB
  __shared__ _Float16 hS[N_GRAPHS * HID];  // 32 KB  (total ~82 KB of 320 KB WGP LDS)
  __shared__ int lenS[N_GRAPHS];
  __shared__ float bihS[192], bhhS[192];

  int tid = threadIdx.x;
  for (int i = tid; i < 192 * 64; i += 1024) {
    WihS[i] = (_Float16)Wih[i];
    WhhS[i] = (_Float16)Whh[i];
  }
  for (int i = tid; i < N_GRAPHS * HID; i += 1024) hS[i] = (_Float16)0.0f;
  if (tid < N_GRAPHS) lenS[tid] = max(lengths[tid], 1);
  if (tid < 192) { bihS[tid] = bih[tid]; bhhS[tid] = bhh[tid]; }
  __syncthreads();

  int wave = tid >> 5, lane = tid & 31;
  int gb  = wave >> 1;
  int jb0 = (wave & 1) * 2;
  int m = lane & 15, half = lane >> 4;
  int gA = gb * 16 + m;   // A-fragment row (graph index)
  int ncol = lane & 15;   // D-fragment column within tile

  float hreg[2][8];
#pragma unroll
  for (int u = 0; u < 2; ++u)
#pragma unroll
    for (int r = 0; r < 8; ++r) hreg[u][r] = 0.0f;

#pragma unroll 1
  for (int t = 0; t < SEQ_L; ++t) {
    // ---- A fragments: x_t (embedding, padding_idx=0 -> zeros) and h (LDS f16)
    int tok = seq[(size_t)gA * SEQ_L + t];
    const float* erow = emb + (size_t)tok * 64;
    v16h ax[2], ah[2];
#pragma unroll
    for (int kc = 0; kc < 2; ++kc) {
#pragma unroll
      for (int i = 0; i < 16; ++i) {
        int k = kc * 32 + half * 16 + i;
        ax[kc][i] = (tok == 0) ? (_Float16)0.0f : (_Float16)erow[k];
        ah[kc][i] = hS[gA * 64 + k];
      }
    }
    // ---- GEMM accumulation (B fragments from LDS-resident f16 weights)
    v8f acc_r[2], acc_z[2], acc_in[2], acc_hn[2];
#pragma unroll
    for (int u = 0; u < 2; ++u) {
      acc_r[u] = vzero8(); acc_z[u] = vzero8();
      acc_in[u] = vzero8(); acc_hn[u] = vzero8();
      int jb = jb0 + u;
      int nr = 0 * 64 + jb * 16 + ncol;
      int nz = 1 * 64 + jb * 16 + ncol;
      int nn = 2 * 64 + jb * 16 + ncol;
#pragma unroll
      for (int kc = 0; kc < 2; ++kc) {
        int kb = kc * 32 + half * 16;
        v16h b;
#pragma unroll
        for (int i = 0; i < 16; ++i) b[i] = WihS[nr * 64 + kb + i];
        acc_r[u] = wmma_f16(ax[kc], b, acc_r[u]);
#pragma unroll
        for (int i = 0; i < 16; ++i) b[i] = WhhS[nr * 64 + kb + i];
        acc_r[u] = wmma_f16(ah[kc], b, acc_r[u]);
#pragma unroll
        for (int i = 0; i < 16; ++i) b[i] = WihS[nz * 64 + kb + i];
        acc_z[u] = wmma_f16(ax[kc], b, acc_z[u]);
#pragma unroll
        for (int i = 0; i < 16; ++i) b[i] = WhhS[nz * 64 + kb + i];
        acc_z[u] = wmma_f16(ah[kc], b, acc_z[u]);
#pragma unroll
        for (int i = 0; i < 16; ++i) b[i] = WihS[nn * 64 + kb + i];
        acc_in[u] = wmma_f16(ax[kc], b, acc_in[u]);
#pragma unroll
        for (int i = 0; i < 16; ++i) b[i] = WhhS[nn * 64 + kb + i];
        acc_hn[u] = wmma_f16(ah[kc], b, acc_hn[u]);
      }
    }
    __syncthreads();  // all hS reads for step t done before overwrite
    // ---- gates + state update (lane-local; exclusive ownership of (g,col))
#pragma unroll
    for (int u = 0; u < 2; ++u) {
      int jb  = jb0 + u;
      int col = jb * 16 + ncol;
      float br_  = bihS[col] + bhhS[col];
      float bz_  = bihS[64 + col] + bhhS[64 + col];
      float bin_ = bihS[128 + col];
      float bhn_ = bhhS[128 + col];
#pragma unroll
      for (int r = 0; r < 8; ++r) {
        int g = gb * 16 + 8 * half + r;
        float rg = 1.0f / (1.0f + expf(-(acc_r[u][r] + br_)));
        float zg = 1.0f / (1.0f + expf(-(acc_z[u][r] + bz_)));
        float ng = tanhf(acc_in[u][r] + bin_ + rg * (acc_hn[u][r] + bhn_));
        float hprev = hreg[u][r];
        float hnew  = (1.0f - zg) * ng + zg * hprev;
        float hcur  = (t < lenS[g]) ? hnew : hprev;  // freeze past true length
        hreg[u][r] = hcur;
        hS[g * 64 + col] = (_Float16)hcur;
      }
    }
    __syncthreads();
  }
  // ---- write final hidden state (f32 master copy)
#pragma unroll
  for (int u = 0; u < 2; ++u) {
    int col = (jb0 + u) * 16 + ncol;
#pragma unroll
    for (int r = 0; r < 8; ++r) {
      int g = gb * 16 + 8 * half + r;
      hT[(size_t)g * 64 + col] = hreg[u][r];
    }
  }
}

// ---------------------------------------------------------------- MLP head (tiny)
__global__ void head_kernel(const float* __restrict__ pooled, const float* __restrict__ hT,
                            const float* __restrict__ Wl1, const float* __restrict__ bl1,
                            const float* __restrict__ Wl2, const float* __restrict__ bl2,
                            const float* __restrict__ Wo, const float* __restrict__ bo,
                            float* __restrict__ out) {
  int g = blockIdx.x * blockDim.x + threadIdx.x;
  if (g >= N_GRAPHS) return;
  float z[128];
  for (int k = 0; k < 64; ++k) { z[k] = pooled[g * 64 + k]; z[64 + k] = hT[g * 64 + k]; }
  float z1[64];
  for (int j = 0; j < 64; ++j) {
    float s = bl1[j];
    for (int k = 0; k < 128; ++k) s += Wl1[j * 128 + k] * z[k];
    z1[j] = fmaxf(s, 0.0f);
  }
  float z2[64];
  for (int j = 0; j < 64; ++j) {
    float s = bl2[j];
    for (int k = 0; k < 64; ++k) s += Wl2[j * 64 + k] * z1[k];
    z2[j] = fmaxf(s, 0.0f);
  }
  float s = bo[0];
  for (int k = 0; k < 64; ++k) s += Wo[k] * z2[k];
  out[g] = s;
}

// ---------------------------------------------------------------- launcher
extern "C" void kernel_launch(void* const* d_in, const int* in_sizes, int n_in,
                              void* d_out, int out_size, void* d_ws, size_t ws_size,
                              hipStream_t stream) {
  const float* x   = (const float*)d_in[0];
  const int*   ei  = (const int*)d_in[1];
  const int* batch = (const int*)d_in[2];
  const int*  seq  = (const int*)d_in[3];
  const int* lens  = (const int*)d_in[4];
  const float* W1r = (const float*)d_in[5];
  const float* b1r = (const float*)d_in[6];
  const float* W1t = (const float*)d_in[7];
  const float* W2r = (const float*)d_in[8];
  const float* b2r = (const float*)d_in[9];
  const float* W2t = (const float*)d_in[10];
  const float* W3r = (const float*)d_in[11];
  const float* b3r = (const float*)d_in[12];
  const float* W3t = (const float*)d_in[13];
  const float* emb = (const float*)d_in[14];
  const float* Wih = (const float*)d_in[15];
  const float* Whh = (const float*)d_in[16];
  const float* bih = (const float*)d_in[17];
  const float* bhh = (const float*)d_in[18];
  const float* Wl1 = (const float*)d_in[19];
  const float* bl1 = (const float*)d_in[20];
  const float* Wl2 = (const float*)d_in[21];
  const float* bl2 = (const float*)d_in[22];
  const float* Wo  = (const float*)d_in[23];
  const float* bo  = (const float*)d_in[24];
  float* out = (float*)d_out;

  const int* esrc = ei;
  const int* edst = ei + N_EDGES;

  float* base   = (float*)d_ws;
  float* deg    = base;             // 50000 (padded to 50176)
  float* cnt    = base + 50176;     // 256
  float* pooled = base + 50432;     // 16384
  float* hT     = base + 66816;     // 16384
  float* agg    = base + 83200;     // 3,200,000
  float* hA     = base + 3283200;   // 3,200,000
  float* hB     = base + 6483200;   // 3,200,000  (total ~38.7 MB)

  hipMemsetAsync(base, 0, (size_t)66816 * 4, stream);  // deg, cnt, pooled
  const int tE = (N_EDGES + 255) / 256;
  deg_kernel<<<tE, 256, 0, stream>>>(edst, deg, N_EDGES);
  cnt_kernel<<<(N_NODES + 255) / 256, 256, 0, stream>>>(batch, cnt, N_NODES);

  const int gw = ((N_NODES / 16) * 32 + 255) / 256;  // 3125 waves -> 391 blocks

  // layer 1
  hipMemsetAsync(agg, 0, (size_t)N_NODES * 64 * 4, stream);
  scatter_kernel<<<tE, 256, 0, stream>>>(x, esrc, edst, agg, N_EDGES);
  gconv_kernel<<<gw, 256, 0, stream>>>(x, agg, deg, W1r, b1r, W1t, hA, 1);
  // layer 2
  hipMemsetAsync(agg, 0, (size_t)N_NODES * 64 * 4, stream);
  scatter_kernel<<<tE, 256, 0, stream>>>(hA, esrc, edst, agg, N_EDGES);
  gconv_kernel<<<gw, 256, 0, stream>>>(hA, agg, deg, W2r, b2r, W2t, hB, 1);
  // layer 3 (no relu)
  hipMemsetAsync(agg, 0, (size_t)N_NODES * 64 * 4, stream);
  scatter_kernel<<<tE, 256, 0, stream>>>(hB, esrc, edst, agg, N_EDGES);
  gconv_kernel<<<gw, 256, 0, stream>>>(hB, agg, deg, W3r, b3r, W3t, hA, 0);

  pool_scatter_kernel<<<(N_NODES + 255) / 256, 256, 0, stream>>>(hA, batch, pooled, N_NODES);
  pool_div_kernel<<<(N_GRAPHS * 64) / 256, 256, 0, stream>>>(pooled, cnt);

  gru_kernel<<<1, 1024, 0, stream>>>(seq, lens, emb, Wih, Whh, bih, bhh, hT);
  head_kernel<<<1, 256, 0, stream>>>(pooled, hT, Wl1, bl1, Wl2, bl2, Wo, bo, out);
}
